// Hello_10694468567054
// MI455X (gfx1250) — compile-verified, matrix operand order checked by reading, not run on hardware
//
#include <hip/hip_runtime.h>
#include <hip/hip_bf16.h>

// ---------------------------------------------------------------------------
// Types for CDNA5 WMMA (wave32): A/B fragments = 16 f16 per lane, C/D = 8 f32.
// ---------------------------------------------------------------------------
typedef __attribute__((ext_vector_type(16))) _Float16 v16h;
typedef __attribute__((ext_vector_type(8)))  float    v8f;

union FragH { v16h h; uint4 q[2]; };

#define EPSV 1e-4f
#define NEGINF (-3.402823466e38f)

// ---------------------------------------------------------------------------
// Zero-fill (d_ws is poisoned by harness; graph capture forbids hipMemset)
// ---------------------------------------------------------------------------
__global__ void zero_kernel(unsigned int* __restrict__ p, long n) {
  long i = (long)blockIdx.x * blockDim.x + threadIdx.x;
  long stride = (long)gridDim.x * blockDim.x;
  for (; i < n; i += stride) p[i] = 0u;
}

// ---------------------------------------------------------------------------
// Scatter: features -> dense f32 grid (B,15,15,15,1) + occupancy mask
// ---------------------------------------------------------------------------
__global__ void scatter_kernel(const float* __restrict__ feat,
                               const int* __restrict__ coords,
                               const int* __restrict__ bidx,
                               float* __restrict__ grid,
                               unsigned char* __restrict__ mask, int N) {
  int i = blockIdx.x * blockDim.x + threadIdx.x;
  if (i >= N) return;
  int b = bidx[i];
  int x = coords[3 * i + 0];
  int y = coords[3 * i + 1];
  int z = coords[3 * i + 2];
  int v = ((b * 15 + x) * 15 + y) * 15 + z;
  atomicAdd(&grid[v], feat[i]);
  mask[v] = 1;
}

// ---------------------------------------------------------------------------
// Layer 0: 1 -> 8 channels, 15^3, SAME. K=27 too skinny for matrix cores:
// scalar conv with weights staged in LDS. Writes f16 NDHWC.
// ---------------------------------------------------------------------------
__global__ void conv0_kernel(const float* __restrict__ grid,
                             const float* __restrict__ w0,
                             _Float16* __restrict__ out, int Mtot) {
  __shared__ float ws0[27 * 8];
  for (int i = threadIdx.x; i < 27 * 8; i += blockDim.x) ws0[i] = w0[i];
  __syncthreads();
  int m = blockIdx.x * blockDim.x + threadIdx.x;
  if (m >= Mtot) return;
  const int D3 = 15 * 15 * 15;
  int b = m / D3;
  int rem = m - b * D3;
  int x = rem / 225, y = (rem / 15) % 15, z = rem % 15;
  float s[8] = {0.f, 0.f, 0.f, 0.f, 0.f, 0.f, 0.f, 0.f};
  #pragma unroll
  for (int tap = 0; tap < 27; ++tap) {
    int nx = x + tap / 9 - 1;
    int ny = y + (tap / 3) % 3 - 1;
    int nz = z + tap % 3 - 1;
    if ((unsigned)nx < 15u && (unsigned)ny < 15u && (unsigned)nz < 15u) {
      float v = grid[((b * 15 + nx) * 15 + ny) * 15 + nz];
      #pragma unroll
      for (int c = 0; c < 8; ++c) s[c] += v * ws0[tap * 8 + c];
    }
  }
  #pragma unroll
  for (int c = 0; c < 8; ++c) out[(size_t)m * 8 + c] = (_Float16)s[c];
}

// ---------------------------------------------------------------------------
// Weight prepack: (3,3,3,Cin,Cout) f32  ->  [Cout][Kpad] f16  (B^T layout so
// B-fragment loads are contiguous 16B; pad rows/K pre-zeroed by zero_kernel).
// ---------------------------------------------------------------------------
template <int CIN, int COUT>
__global__ void prepack_kernel(const float* __restrict__ w, _Float16* __restrict__ wp) {
  constexpr int KTOT = 27 * CIN;
  constexpr int KPAD = ((KTOT + 31) / 32) * 32;
  int i = blockIdx.x * blockDim.x + threadIdx.x;
  if (i < KTOT * COUT) {
    int co = i % COUT;
    int k  = i / COUT;            // k = tap*CIN + ci  (matches w flat layout)
    wp[(size_t)co * KPAD + k] = (_Float16)w[i];
  }
}

// ---------------------------------------------------------------------------
// Implicit-GEMM conv via V_WMMA_F32_16X16X32_F16.
// A fragment per-lane addressing follows the 16-bit A-matrix 16x32 layout:
//   lane 0-15  (khalf=0): K in {kb..kb+7} U {kb+16..kb+23}
//   lane 16-31 (khalf=1): shifted by 8.
// Each 8-wide K-group is contiguous in NDHWC memory (Cin % 8 == 0), so one
// guarded 16-byte global load per group (zero for OOB taps / padded K).
// K loop fully unrolled (KSTEPS = 7/14/21) so loads clause + overlap WMMA.
// ---------------------------------------------------------------------------
template <int CIN, int DD>
__device__ __forceinline__ uint4 a_group(const _Float16* __restrict__ act,
                                         int bi, int x, int y, int z, int kb) {
  constexpr int KTOT = 27 * CIN;
  uint4 r = {0u, 0u, 0u, 0u};
  if (kb < KTOT) {
    int tap = kb / CIN;
    int co  = kb - tap * CIN;
    int nx = x + tap / 9 - 1;
    int ny = y + (tap / 3) % 3 - 1;
    int nz = z + tap % 3 - 1;
    if ((unsigned)nx < (unsigned)DD && (unsigned)ny < (unsigned)DD &&
        (unsigned)nz < (unsigned)DD) {
      size_t idx = ((((size_t)bi * DD + nx) * DD + ny) * DD + nz) * CIN + co;
      r = *reinterpret_cast<const uint4*>(act + idx);   // 16B aligned: idx%8==0
    }
  }
  return r;
}

template <int CIN, int COUT, int DD>
__global__ void conv_wmma_kernel(const _Float16* __restrict__ act,
                                 const _Float16* __restrict__ wp,
                                 _Float16* __restrict__ out, int Mtot) {
  constexpr int NT     = (COUT + 15) / 16;
  constexpr int KSTEPS = (27 * CIN + 31) / 32;
  constexpr int KPAD   = KSTEPS * 32;
  constexpr int D3     = DD * DD * DD;

  int wave = (blockIdx.x * blockDim.x + threadIdx.x) >> 5;   // wave32
  int lane = threadIdx.x & 31;
  int mtile = wave;
  if (mtile * 16 >= Mtot) return;                            // wave-uniform

  int row   = lane & 15;
  int khalf = lane >> 4;
  int m  = mtile * 16 + row;
  int bi = m / D3;
  int rem = m - bi * D3;
  int x = rem / (DD * DD);
  int y = (rem / DD) % DD;
  int z = rem % DD;

  v8f acc[NT] = {};

  const _Float16* wr = wp + (size_t)(lane & 15) * KPAD;
  __builtin_prefetch(wr, 0, 3);   // global_prefetch_b8 of this lane's B row

  #pragma unroll
  for (int ks = 0; ks < KSTEPS; ++ks) {
    int kb = ks * 32 + khalf * 8;
    FragH a;
    a.q[0] = a_group<CIN, DD>(act, bi, x, y, z, kb);
    a.q[1] = a_group<CIN, DD>(act, bi, x, y, z, kb + 16);
    #pragma unroll
    for (int t = 0; t < NT; ++t) {
      FragH bfr;
      const _Float16* wrow = wr + (size_t)t * 16 * KPAD;
      bfr.q[0] = *reinterpret_cast<const uint4*>(wrow + kb);
      bfr.q[1] = *reinterpret_cast<const uint4*>(wrow + kb + 16);
      acc[t] = __builtin_amdgcn_wmma_f32_16x16x32_f16(
          /*neg_a=*/false, a.h, /*neg_b=*/false, bfr.h,
          /*c_mod=*/(short)0, acc[t], /*reuse_a=*/false, /*reuse_b=*/false);
    }
  }

  // D layout: lane 0-15 VGPR r -> (M=r, N=lane); lane 16-31 -> (M=r+8, N=lane-16)
  #pragma unroll
  for (int t = 0; t < NT; ++t) {
    int ch = t * 16 + (lane & 15);
    if (ch < COUT) {
      #pragma unroll
      for (int r = 0; r < 8; ++r) {
        int mm = mtile * 16 + r + 8 * khalf;
        out[(size_t)mm * COUT + ch] = (_Float16)acc[t][r];
      }
    }
  }
}

// ---------------------------------------------------------------------------
// Masked BN statistics: stats = [count, sum[C], sumsq[C]] (f32).
// Per-thread register accumulation over grid-stride loop; one LDS-atomic set
// per thread and one global-atomic set per block (vs per-voxel atomics).
// Channel rows loaded as 16B vectors (C % 8 == 0, rows 16B-aligned).
// ---------------------------------------------------------------------------
template <int C>
__global__ void bn_reduce_kernel(const _Float16* __restrict__ y,
                                 const unsigned char* __restrict__ mask,
                                 float* __restrict__ stats, int Mvox) {
  float cnt = 0.f;
  float s1[C] = {};
  float s2[C] = {};
  for (int m = blockIdx.x * blockDim.x + threadIdx.x; m < Mvox;
       m += gridDim.x * blockDim.x) {
    if (mask[m]) {
      cnt += 1.f;
      union { uint4 q[C / 8]; _Float16 h[C]; } r;
      const _Float16* rowp = y + (size_t)m * C;
      #pragma unroll
      for (int j = 0; j < C / 8; ++j)
        r.q[j] = *reinterpret_cast<const uint4*>(rowp + 8 * j);
      #pragma unroll
      for (int c = 0; c < C; ++c) {
        float v = (float)r.h[c];
        s1[c] += v;
        s2[c] += v * v;
      }
    }
  }
  __shared__ float ls[2 * C + 1];
  for (int i = threadIdx.x; i < 2 * C + 1; i += blockDim.x) ls[i] = 0.f;
  __syncthreads();
  atomicAdd(&ls[0], cnt);
  #pragma unroll
  for (int c = 0; c < C; ++c) {
    atomicAdd(&ls[1 + c], s1[c]);
    atomicAdd(&ls[1 + C + c], s2[c]);
  }
  __syncthreads();
  for (int i = threadIdx.x; i < 2 * C + 1; i += blockDim.x)
    atomicAdd(&stats[i], ls[i]);
}

// ---------------------------------------------------------------------------
// Masked BN apply + ReLU + mask (in place, f16, vectorized rows)
// ---------------------------------------------------------------------------
template <int C>
__global__ void bn_apply_kernel(_Float16* __restrict__ y,
                                const unsigned char* __restrict__ mask,
                                const float* __restrict__ stats,
                                const float* __restrict__ g,
                                const float* __restrict__ bt, int Mvox) {
  int m = blockIdx.x * blockDim.x + threadIdx.x;
  if (m >= Mvox) return;
  float n = fmaxf(stats[0], 1.0f);
  bool mk = mask[m] != 0;
  union { uint4 q[C / 8]; _Float16 h[C]; } r;
  _Float16* rowp = y + (size_t)m * C;
  #pragma unroll
  for (int j = 0; j < C / 8; ++j)
    r.q[j] = *reinterpret_cast<const uint4*>(rowp + 8 * j);
  #pragma unroll
  for (int c = 0; c < C; ++c) {
    float mean = stats[1 + c] / n;
    float var  = stats[1 + C + c] / n - mean * mean;
    float v = (float)r.h[c];
    float o = fmaxf((v - mean) * rsqrtf(var + EPSV) * g[c] + bt[c], 0.0f);
    r.h[c] = (_Float16)(mk ? o : 0.0f);
  }
  #pragma unroll
  for (int j = 0; j < C / 8; ++j)
    *reinterpret_cast<uint4*>(rowp + 8 * j) = r.q[j];
}

// Final layer: BN+ReLU+mask, write f32 output (B,32,1,1,1) -> flat b*32+c
__global__ void bn_final_kernel(const _Float16* __restrict__ y,
                                const unsigned char* __restrict__ mask,
                                const float* __restrict__ stats,
                                const float* __restrict__ g,
                                const float* __restrict__ bt,
                                float* __restrict__ out, int Bn) {
  int i = blockIdx.x * blockDim.x + threadIdx.x;
  if (i >= Bn * 32) return;
  int m = i >> 5;
  int c = i & 31;
  float n = fmaxf(stats[0], 1.0f);
  float mean = stats[1 + c] / n;
  float var  = stats[1 + 32 + c] / n - mean * mean;
  float v = (float)y[(size_t)m * 32 + c];
  float o = fmaxf((v - mean) * rsqrtf(var + EPSV) * g[c] + bt[c], 0.0f);
  out[i] = mask[m] ? o : 0.0f;
}

// ---------------------------------------------------------------------------
// Masked maxpool 3^3 stride 2 VALID + pooled mask (vectorized channel rows)
// ---------------------------------------------------------------------------
template <int C, int DI, int DO>
__global__ void maxpool_kernel(const _Float16* __restrict__ x,
                               const unsigned char* __restrict__ mi,
                               _Float16* __restrict__ o,
                               unsigned char* __restrict__ mo, int Mout) {
  int m = blockIdx.x * blockDim.x + threadIdx.x;
  if (m >= Mout) return;
  constexpr int DO3 = DO * DO * DO;
  int b = m / DO3;
  int rem = m - b * DO3;
  int ox = rem / (DO * DO), oy = (rem / DO) % DO, oz = rem % DO;
  float best[C];
  #pragma unroll
  for (int c = 0; c < C; ++c) best[c] = NEGINF;
  bool any = false;
  #pragma unroll
  for (int dx = 0; dx < 3; ++dx)
    #pragma unroll
    for (int dy = 0; dy < 3; ++dy)
      #pragma unroll
      for (int dz = 0; dz < 3; ++dz) {
        int ix = 2 * ox + dx, iy = 2 * oy + dy, iz = 2 * oz + dz; // always < DI
        int vi = ((b * DI + ix) * DI + iy) * DI + iz;
        if (mi[vi]) {
          any = true;
          union { uint4 q[C / 8]; _Float16 h[C]; } r;
          const _Float16* rowp = x + (size_t)vi * C;
          #pragma unroll
          for (int j = 0; j < C / 8; ++j)
            r.q[j] = *reinterpret_cast<const uint4*>(rowp + 8 * j);
          #pragma unroll
          for (int c = 0; c < C; ++c)
            best[c] = fmaxf(best[c], (float)r.h[c]);
        }
      }
  mo[m] = any ? 1 : 0;
  union { uint4 q[C / 8]; _Float16 h[C]; } w;
  #pragma unroll
  for (int c = 0; c < C; ++c) w.h[c] = (_Float16)(any ? best[c] : 0.0f);
  _Float16* orow = o + (size_t)m * C;
  #pragma unroll
  for (int j = 0; j < C / 8; ++j)
    *reinterpret_cast<uint4*>(orow + 8 * j) = w.q[j];
}

// ---------------------------------------------------------------------------
// Host orchestration (triple-chevron launches)
// ---------------------------------------------------------------------------
static inline int nblk(long n, int bs) { return (int)((n + bs - 1) / bs); }

extern "C" void kernel_launch(void* const* d_in, const int* in_sizes, int n_in,
                              void* d_out, int out_size, void* d_ws, size_t ws_size,
                              hipStream_t stream) {
  const float* feat  = (const float*)d_in[0];
  const int* coords  = (const int*)d_in[1];
  const int* bidx    = (const int*)d_in[2];
  const float* w[7]; const float* g[7]; const float* bt[7];
  for (int i = 0; i < 7; ++i) {
    w[i]  = (const float*)d_in[3 + 3 * i];
    g[i]  = (const float*)d_in[4 + 3 * i];
    bt[i] = (const float*)d_in[5 + 3 * i];
  }
  float* out = (float*)d_out;

  const int N  = in_sizes[0];                 // 524288
  const int M0 = 1024 * 15 * 15 * 15;         // 3,456,000 voxels
  const int M1 = 1024 * 7 * 7 * 7;            // 351,232
  const int M2 = 1024 * 3 * 3 * 3;            // 27,648
  const int M3 = 1024;                        // 1,024

  // ---- scratch partition (all 256B aligned) ----
  size_t off = 0;
  auto take = [&](size_t bytes) -> void* {
    off = (off + 255) & ~(size_t)255;
    void* p = (char*)d_ws + off;
    off += bytes;
    return p;
  };
  float*         grid0 = (float*)take((size_t)M0 * 4);
  unsigned char* mask0 = (unsigned char*)take(M0);
  unsigned char* mask1 = (unsigned char*)take(M1);
  unsigned char* mask2 = (unsigned char*)take(M2);
  unsigned char* mask3 = (unsigned char*)take(M3);
  _Float16*      actA  = (_Float16*)take((size_t)M0 * 8 * 2);
  _Float16*      actB  = (_Float16*)take((size_t)M0 * 8 * 2);
  _Float16*      wpack = (_Float16*)take(32 * 672 * 2);    // max [32][672] f16
  float*         stats = (float*)take(256);                // [1 + 2*32] f32

  const int BS = 256;
  const long WPWORDS = 32 * 672 * 2 / 4;   // wpack in u32 words
  const long STWORDS = 65;                 // stats words

  auto zero = [&](void* p, long words) {
    int b = nblk(words, BS); if (b > 2048) b = 2048;
    zero_kernel<<<dim3(b), dim3(BS), 0, stream>>>((unsigned int*)p, words);
  };

  // ---- scatter to dense grid ----
  zero(grid0, (long)M0);
  zero(mask0, (long)M0 / 4);
  scatter_kernel<<<dim3(nblk(N, BS)), dim3(BS), 0, stream>>>(
      feat, coords, bidx, grid0, mask0, N);

  // ---- layer 0: conv 1->8 (VALU) + BN ----
  conv0_kernel<<<dim3(nblk(M0, BS)), dim3(BS), 0, stream>>>(grid0, w[0], actA, M0);
  zero(stats, STWORDS);
  bn_reduce_kernel<8><<<dim3(2048), dim3(BS), 0, stream>>>(actA, mask0, stats, M0);
  bn_apply_kernel<8><<<dim3(nblk(M0, BS)), dim3(BS), 0, stream>>>(
      actA, mask0, stats, g[0], bt[0], M0);

  // ---- layer 1: conv 8->8 WMMA + BN ----
  zero(wpack, WPWORDS);
  prepack_kernel<8, 8><<<dim3(nblk(27 * 8 * 8, BS)), dim3(BS), 0, stream>>>(w[1], wpack);
  conv_wmma_kernel<8, 8, 15><<<dim3((M0 / 16) * 32 / BS), dim3(BS), 0, stream>>>(
      actA, wpack, actB, M0);
  zero(stats, STWORDS);
  bn_reduce_kernel<8><<<dim3(2048), dim3(BS), 0, stream>>>(actB, mask0, stats, M0);
  bn_apply_kernel<8><<<dim3(nblk(M0, BS)), dim3(BS), 0, stream>>>(
      actB, mask0, stats, g[1], bt[1], M0);

  // ---- pool 15 -> 7 ----
  maxpool_kernel<8, 15, 7><<<dim3(nblk(M1, BS)), dim3(BS), 0, stream>>>(
      actB, mask0, actA, mask1, M1);

  // ---- layer 2: conv 8->16 WMMA + BN ----
  zero(wpack, WPWORDS);
  prepack_kernel<8, 16><<<dim3(nblk(27 * 8 * 16, BS)), dim3(BS), 0, stream>>>(w[2], wpack);
  conv_wmma_kernel<8, 16, 7><<<dim3((M1 / 16) * 32 / BS), dim3(BS), 0, stream>>>(
      actA, wpack, actB, M1);
  zero(stats, STWORDS);
  bn_reduce_kernel<16><<<dim3(1024), dim3(BS), 0, stream>>>(actB, mask1, stats, M1);
  bn_apply_kernel<16><<<dim3(nblk(M1, BS)), dim3(BS), 0, stream>>>(
      actB, mask1, stats, g[2], bt[2], M1);

  // ---- layer 3: conv 16->16 WMMA + BN ----
  zero(wpack, WPWORDS);
  prepack_kernel<16, 16><<<dim3(nblk(27 * 16 * 16, BS)), dim3(BS), 0, stream>>>(w[3], wpack);
  conv_wmma_kernel<16, 16, 7><<<dim3((M1 / 16) * 32 / BS), dim3(BS), 0, stream>>>(
      actB, wpack, actA, M1);
  zero(stats, STWORDS);
  bn_reduce_kernel<16><<<dim3(1024), dim3(BS), 0, stream>>>(actA, mask1, stats, M1);
  bn_apply_kernel<16><<<dim3(nblk(M1, BS)), dim3(BS), 0, stream>>>(
      actA, mask1, stats, g[3], bt[3], M1);

  // ---- pool 7 -> 3 ----
  maxpool_kernel<16, 7, 3><<<dim3(nblk(M2, BS)), dim3(BS), 0, stream>>>(
      actA, mask1, actB, mask2, M2);

  // ---- layer 4: conv 16->24 WMMA + BN ----
  zero(wpack, WPWORDS);
  prepack_kernel<16, 24><<<dim3(nblk(27 * 16 * 24, BS)), dim3(BS), 0, stream>>>(w[4], wpack);
  conv_wmma_kernel<16, 24, 3><<<dim3((M2 / 16) * 32 / BS), dim3(BS), 0, stream>>>(
      actB, wpack, actA, M2);
  zero(stats, STWORDS);
  bn_reduce_kernel<24><<<dim3(108), dim3(BS), 0, stream>>>(actA, mask2, stats, M2);
  bn_apply_kernel<24><<<dim3(nblk(M2, BS)), dim3(BS), 0, stream>>>(
      actA, mask2, stats, g[4], bt[4], M2);

  // ---- layer 5: conv 24->24 WMMA + BN ----
  zero(wpack, WPWORDS);
  prepack_kernel<24, 24><<<dim3(nblk(27 * 24 * 24, BS)), dim3(BS), 0, stream>>>(w[5], wpack);
  conv_wmma_kernel<24, 24, 3><<<dim3((M2 / 16) * 32 / BS), dim3(BS), 0, stream>>>(
      actA, wpack, actB, M2);
  zero(stats, STWORDS);
  bn_reduce_kernel<24><<<dim3(108), dim3(BS), 0, stream>>>(actB, mask2, stats, M2);
  bn_apply_kernel<24><<<dim3(nblk(M2, BS)), dim3(BS), 0, stream>>>(
      actB, mask2, stats, g[5], bt[5], M2);

  // ---- pool 3 -> 1 ----
  maxpool_kernel<24, 3, 1><<<dim3(nblk(M3, BS)), dim3(BS), 0, stream>>>(
      actB, mask2, actA, mask3, M3);

  // ---- layer 6: conv 24->32 WMMA + BN -> f32 output ----
  zero(wpack, WPWORDS);
  prepack_kernel<24, 32><<<dim3(nblk(27 * 24 * 32, BS)), dim3(BS), 0, stream>>>(w[6], wpack);
  conv_wmma_kernel<24, 32, 1><<<dim3((M3 / 16) * 32 / BS), dim3(BS), 0, stream>>>(
      actA, wpack, actB, M3);
  zero(stats, STWORDS);
  bn_reduce_kernel<32><<<dim3(4), dim3(BS), 0, stream>>>(actB, mask3, stats, M3);
  bn_final_kernel<<<dim3(nblk(M3 * 32, BS)), dim3(BS), 0, stream>>>(
      actB, mask3, stats, g[6], bt[6], out, M3);
}
